// GNNnetwork_76209899700463
// MI455X (gfx1250) — compile-verified, matrix operand order checked by reading, not run on hardware
//
#include <hip/hip_runtime.h>

typedef __attribute__((ext_vector_type(16))) __bf16 v16bf;
typedef __attribute__((ext_vector_type(8)))  float  v8f;

#define N_NODES 50000
#define N_EDGES 800000
#define DIM     64
#define LAYERS  5
#define NEDGE   4
#define NGRAPH  512
#define NTASK   10
#define NTILES  (N_NODES / 16)   // 3125, exact

// ---------------------------------------------------------------- embed (float4)
__global__ __launch_bounds__(256)
void embed_kernel(const int* __restrict__ x, const float* __restrict__ feat_emb,
                  float* __restrict__ h) {
    int t = blockIdx.x * blockDim.x + threadIdx.x;          // N * 16
    if (t >= N_NODES * (DIM / 4)) return;
    int i = t >> 4, q = (t & 15) << 2;
    *(float4*)&h[i * DIM + q] = *(const float4*)&feat_emb[x[i] * DIM + q];
}

// ------------------------------------------------- edge gather + scatter (float4)
__global__ __launch_bounds__(256)
void message_kernel(const float* __restrict__ h, const int* __restrict__ src,
                    const int* __restrict__ dst, const int* __restrict__ eattr,
                    const float* __restrict__ eemb,  // edge_emb for this layer [NE,DIM]
                    float* __restrict__ agg) {
    int t = blockIdx.x * blockDim.x + threadIdx.x;          // E * 16
    if (t >= N_EDGES * (DIM / 4)) return;
    int e = t >> 4, q = (t & 15) << 2;
    float4 hv = *(const float4*)&h[src[e] * DIM + q];
    float4 ev = *(const float4*)&eemb[eattr[e] * DIM + q];
    float* ag = &agg[dst[e] * DIM + q];
    float m0 = hv.x + ev.x, m1 = hv.y + ev.y, m2 = hv.z + ev.z, m3 = hv.w + ev.w;
    if (m0 > 0.0f) unsafeAtomicAdd(ag + 0, m0);             // relu + skip zeros
    if (m1 > 0.0f) unsafeAtomicAdd(ag + 1, m1);
    if (m2 > 0.0f) unsafeAtomicAdd(ag + 2, m2);
    if (m3 > 0.0f) unsafeAtomicAdd(ag + 3, m3);
}

// ------------------------------------------- WMMA GEMM with BN-stat epilogue
// mode 0: A = (1+eps[layer])*in0 + in1          (GINE combine)
// mode 1: A = relu(scale[k]*in0 + shift[k])     (BN+ReLU folded into A load)
// out = A @ W + bias ; accumulates per-column sum/sumsq for the following BN.
// W is staged in LDS pre-swizzled into WMMA B-fragment order so each lane's
// 16 bf16 B elements are one contiguous 32-byte chunk (2x ds_load_b128).
__global__ __launch_bounds__(256)
void gemm_bn_kernel(const float* __restrict__ in0, const float* __restrict__ in1,
                    const float* __restrict__ scale, const float* __restrict__ shift,
                    const float* __restrict__ epsArr, int layer, int mode,
                    const float* __restrict__ W,     // [64,64] row-major (k,n)
                    const float* __restrict__ bias,  // [64]
                    float* __restrict__ out,         // [N,64]
                    float* __restrict__ colsum, float* __restrict__ colsumsq) {
    __shared__ v16bf wfrag[256];                     // [(c*2+kb)*32 + lane]
    __shared__ float blds[DIM];
    __shared__ float sclds[DIM], shlds[DIM];
    int tid = threadIdx.x;

    // stage swizzled weights: idx -> (c[2b], kb[1b], lane[5b], j[4b])
    __bf16* wl = (__bf16*)wfrag;
    for (int idx = tid; idx < DIM * DIM; idx += blockDim.x) {
        int j  = idx & 15;
        int ln = (idx >> 4) & 31;
        int kb = (idx >> 9) & 1;
        int c  = idx >> 10;
        int k  = kb * 32 + j + (j & 8) + ((ln >> 4) << 3);
        int n  = c * 16 + (ln & 15);
        wl[idx] = (__bf16)W[k * DIM + n];
    }
    if (tid < DIM) {
        blds[tid] = bias[tid];
        if (mode == 1) { sclds[tid] = scale[tid]; shlds[tid] = shift[tid]; }
    }
    __syncthreads();

    int wave = tid >> 5;
    int lane = tid & 31;
    int tile = blockIdx.x * 8 + wave;
    if (tile >= NTILES) return;
    int base = tile * 16;
    int hf   = lane >> 4;
    int r    = lane & 15;
    int row  = base + r;

    // A fragments: per lane two contiguous 8-float runs per K-step -> b128 loads
    v16bf a[2];
    const float4* p0 = (const float4*)(in0 + row * DIM);
    if (mode == 0) {
        const float epsl = 1.0f + epsArr[layer];
        const float4* p1 = (const float4*)(in1 + row * DIM);
#pragma unroll
        for (int kb = 0; kb < 2; ++kb) {
            int f = kb * 8 + hf * 2;                       // float4 idx of run start
            float4 xx[4] = { p0[f], p0[f + 1], p0[f + 4], p0[f + 5] };
            float4 yy[4] = { p1[f], p1[f + 1], p1[f + 4], p1[f + 5] };
#pragma unroll
            for (int q = 0; q < 4; ++q) {
                const float* xv = (const float*)&xx[q];
                const float* yv = (const float*)&yy[q];
#pragma unroll
                for (int e = 0; e < 4; ++e)
                    a[kb][q * 4 + e] = (__bf16)(epsl * xv[e] + yv[e]);
            }
        }
    } else {
#pragma unroll
        for (int kb = 0; kb < 2; ++kb) {
            int f  = kb * 8 + hf * 2;
            int k0 = f * 4;                                 // float idx of run start
            float4 xx[4] = { p0[f], p0[f + 1], p0[f + 4], p0[f + 5] };
            float4 sc[4] = { *(const float4*)&sclds[k0],      *(const float4*)&sclds[k0 + 4],
                             *(const float4*)&sclds[k0 + 16], *(const float4*)&sclds[k0 + 20] };
            float4 sh[4] = { *(const float4*)&shlds[k0],      *(const float4*)&shlds[k0 + 4],
                             *(const float4*)&shlds[k0 + 16], *(const float4*)&shlds[k0 + 20] };
#pragma unroll
            for (int q = 0; q < 4; ++q) {
                const float* xv = (const float*)&xx[q];
                const float* sv = (const float*)&sc[q];
                const float* tv = (const float*)&sh[q];
#pragma unroll
                for (int e = 0; e < 4; ++e) {
                    float u = sv[e] * xv[e] + tv[e];
                    a[kb][q * 4 + e] = (__bf16)(u > 0.0f ? u : 0.0f);
                }
            }
        }
    }

#pragma unroll
    for (int c = 0; c < 4; ++c) {                           // 4 output column tiles
        v8f acc = {};
#pragma unroll
        for (int kb = 0; kb < 2; ++kb) {
            v16bf b = wfrag[(c * 2 + kb) * 32 + lane];      // 2x ds_load_b128
            acc = __builtin_amdgcn_wmma_f32_16x16x32_bf16(
                false, a[kb], false, b, (short)0, acc, false, false);
        }
        int n = c * 16 + r;                                 // lane's output column
        float s = 0.0f, s2 = 0.0f;
#pragma unroll
        for (int i = 0; i < 8; ++i) {
            int m = i + hf * 8;                             // lane's output row
            float val = acc[i] + blds[n];
            out[(base + m) * DIM + n] = val;
            s += val; s2 += val * val;
        }
        atomicAdd(&colsum[n], s);
        atomicAdd(&colsumsq[n], s2);
    }
}

// ------------------------------------------- BN stat finalize (1 block, 64 thr)
__global__ void bnfin_kernel(const float* __restrict__ sum, const float* __restrict__ sumsq,
                             const float* __restrict__ g, const float* __restrict__ be,
                             float* __restrict__ scale, float* __restrict__ shift, float invn) {
    int d = threadIdx.x;
    float mean = sum[d] * invn;
    float var  = sumsq[d] * invn - mean * mean;
    float sc   = g[d] * rsqrtf(var + 1e-5f);
    scale[d] = sc;
    shift[d] = be[d] - mean * sc;
}

// ------------------------------------------- apply outer BN + ReLU -> new h (float4)
__global__ __launch_bounds__(256)
void apply_bn_kernel(const float* __restrict__ v, const float* __restrict__ scale,
                     const float* __restrict__ shift, float* __restrict__ h) {
    int t = blockIdx.x * blockDim.x + threadIdx.x;          // N * 16
    if (t >= N_NODES * (DIM / 4)) return;
    int q = (t & 15) << 2;
    float4 x  = *(const float4*)&v[t << 2];
    float4 sc = *(const float4*)&scale[q];
    float4 sh = *(const float4*)&shift[q];
    float4 o;
    o.x = fmaxf(sc.x * x.x + sh.x, 0.0f);
    o.y = fmaxf(sc.y * x.y + sh.y, 0.0f);
    o.z = fmaxf(sc.z * x.z + sh.z, 0.0f);
    o.w = fmaxf(sc.w * x.w + sh.w, 0.0f);
    *(float4*)&h[t << 2] = o;
}

// ------------------------------------------- global add pool (float4)
__global__ __launch_bounds__(256)
void pool_kernel(const float* __restrict__ h, const int* __restrict__ batch,
                 float* __restrict__ pooled) {
    int t = blockIdx.x * blockDim.x + threadIdx.x;          // N * 16
    if (t >= N_NODES * (DIM / 4)) return;
    int i = t >> 4, q = (t & 15) << 2;
    float4 hv = *(const float4*)&h[i * DIM + q];
    float* pg = &pooled[batch[i] * DIM + q];
    unsafeAtomicAdd(pg + 0, hv.x);
    unsafeAtomicAdd(pg + 1, hv.y);
    unsafeAtomicAdd(pg + 2, hv.z);
    unsafeAtomicAdd(pg + 3, hv.w);
}

// ------------------------------------------- tiny MLP head (compute-trivial)
__global__ __launch_bounds__(256)
void mlp1_kernel(const float* __restrict__ pooled, const float* __restrict__ Wf1,
                 const float* __restrict__ bf1, float* __restrict__ hid) {
    int t = blockIdx.x * blockDim.x + threadIdx.x;
    if (t >= NGRAPH * 2 * DIM) return;
    int g = t >> 7, j = t & 127;
    float acc = bf1[j];
#pragma unroll 8
    for (int k = 0; k < DIM; ++k) acc += pooled[g * DIM + k] * Wf1[k * 2 * DIM + j];
    hid[t] = acc > 0.0f ? acc : 0.0f;
}

__global__ __launch_bounds__(64)
void mlp2_kernel(const float* __restrict__ hid, const float* __restrict__ Wf2,
                 const float* __restrict__ bf2, float* __restrict__ out) {
    int t = blockIdx.x * blockDim.x + threadIdx.x;
    if (t >= NGRAPH * NTASK) return;
    int g = t / NTASK, j = t % NTASK;
    float acc = bf2[j];
#pragma unroll 8
    for (int k = 0; k < 2 * DIM; ++k) acc += hid[g * 2 * DIM + k] * Wf2[k * NTASK + j];
    out[t] = acc;
}

// ==================================================================== launch
extern "C" void kernel_launch(void* const* d_in, const int* in_sizes, int n_in,
                              void* d_out, int out_size, void* d_ws, size_t ws_size,
                              hipStream_t stream) {
    (void)in_sizes; (void)n_in; (void)out_size; (void)ws_size;
    const int*   x        = (const int*)d_in[0];
    const int*   eidx     = (const int*)d_in[1];    // [2,E]
    const int*   eattr    = (const int*)d_in[2];
    const int*   batch    = (const int*)d_in[3];
    const float* feat_emb = (const float*)d_in[4];
    const float* edge_emb = (const float*)d_in[5];  // [L,NE,D]
    const float* epsArr   = (const float*)d_in[6];
    const float* W1 = (const float*)d_in[7];
    const float* b1 = (const float*)d_in[8];
    const float* g1 = (const float*)d_in[9];
    const float* be1 = (const float*)d_in[10];
    const float* W2 = (const float*)d_in[11];
    const float* b2 = (const float*)d_in[12];
    const float* g2 = (const float*)d_in[13];
    const float* be2 = (const float*)d_in[14];
    const float* Wf1 = (const float*)d_in[15];
    const float* bf1 = (const float*)d_in[16];
    const float* Wf2 = (const float*)d_in[17];
    const float* bf2 = (const float*)d_in[18];

    float* ws     = (float*)d_ws;
    float* h      = ws;
    float* bufA   = h    + (size_t)N_NODES * DIM;   // agg, then v
    float* bufB   = bufA + (size_t)N_NODES * DIM;   // u
    float* pooled = bufB + (size_t)N_NODES * DIM;
    float* hid    = pooled + NGRAPH * DIM;
    float* stats  = hid + NGRAPH * 2 * DIM;         // 512 floats
    float* sum1 = stats;       float* sq1 = stats + 64;
    float* sc1  = stats + 128; float* sh1 = stats + 192;
    float* sum2 = stats + 256; float* sq2 = stats + 320;
    float* sc2  = stats + 384; float* sh2 = stats + 448;

    const int* src = eidx;
    const int* dst = eidx + N_EDGES;

    const int nodeVec   = N_NODES * (DIM / 4);              // 800K float4 slots
    const int nodeVecBl = (nodeVec + 255) / 256;
    const int edgeVecBl = (N_EDGES * (DIM / 4) + 255) / 256; // 50000
    const int gemmBl    = (NTILES + 7) / 8;                  // 391
    const float invn = 1.0f / (float)N_NODES;

    embed_kernel<<<nodeVecBl, 256, 0, stream>>>(x, feat_emb, h);

    for (int l = 0; l < LAYERS; ++l) {
        hipMemsetAsync(bufA, 0, (size_t)N_NODES * DIM * sizeof(float), stream);
        hipMemsetAsync(stats, 0, 512 * sizeof(float), stream);

        message_kernel<<<edgeVecBl, 256, 0, stream>>>(
            h, src, dst, eattr, edge_emb + (size_t)l * NEDGE * DIM, bufA);

        // u = ((1+eps)h + agg) @ W1 + b1 ; BN1 stats
        gemm_bn_kernel<<<gemmBl, 256, 0, stream>>>(
            h, bufA, nullptr, nullptr, epsArr, l, 0,
            W1 + (size_t)l * DIM * DIM, b1 + l * DIM, bufB, sum1, sq1);
        bnfin_kernel<<<1, 64, 0, stream>>>(sum1, sq1, g1 + l * DIM, be1 + l * DIM,
                                           sc1, sh1, invn);
        // v = relu(BN1(u)) @ W2 + b2 ; BN2 stats
        gemm_bn_kernel<<<gemmBl, 256, 0, stream>>>(
            bufB, nullptr, sc1, sh1, epsArr, l, 1,
            W2 + (size_t)l * DIM * DIM, b2 + l * DIM, bufA, sum2, sq2);
        bnfin_kernel<<<1, 64, 0, stream>>>(sum2, sq2, g2 + l * DIM, be2 + l * DIM,
                                           sc2, sh2, invn);
        // h = relu(BN2(v))
        apply_bn_kernel<<<nodeVecBl, 256, 0, stream>>>(bufA, sc2, sh2, h);
    }

    hipMemsetAsync(pooled, 0, NGRAPH * DIM * sizeof(float), stream);
    pool_kernel<<<nodeVecBl, 256, 0, stream>>>(h, batch, pooled);
    mlp1_kernel<<<(NGRAPH * 2 * DIM + 255) / 256, 256, 0, stream>>>(pooled, Wf1, bf1, hid);
    mlp2_kernel<<<(NGRAPH * NTASK + 63) / 64, 64, 0, stream>>>(hid, Wf2, bf2, (float*)d_out);
}
